// ScaledDotSelfAttention_62045097558233
// MI455X (gfx1250) — compile-verified, matrix operand order checked by reading, not run on hardware
//
#include <hip/hip_runtime.h>
#include <hip/hip_bf16.h>

typedef __attribute__((ext_vector_type(16))) _Float16 v16h;
typedef __attribute__((ext_vector_type(8)))  _Float16 v8h;
typedef __attribute__((ext_vector_type(4)))  _Float16 v4h;
typedef __attribute__((ext_vector_type(8)))  float    v8f;

// ---------------------------------------------------------------------------
// CDNA5 async global->LDS copy (per-lane, 16 bytes), tracked by ASYNCcnt.
// Inline asm per CDNA5 ISA: VDST = LDS byte address (the
// low 32 bits of a generic LDS pointer), VADDR = 64-bit global address.
// ---------------------------------------------------------------------------
__device__ __forceinline__ void async_copy_b128(const void* gsrc, void* ldst)
{
    const unsigned           lds = (unsigned)(__UINTPTR_TYPE__)ldst;
    const unsigned long long ga  = (unsigned long long)(__UINTPTR_TYPE__)gsrc;
    asm volatile("global_load_async_to_lds_b128 %0, %1, off"
                 :: "v"(lds), "v"(ga) : "memory");
}

// Drain this wave's ASYNCcnt, then workgroup barrier: the required protocol
// before other waves may read async-staged LDS data. No-op wait when no
// async ops were issued (f32-source instantiations).
__device__ __forceinline__ void wait_async_then_barrier()
{
    asm volatile("s_wait_asynccnt 0x0" ::: "memory");
    __syncthreads();
}

// ---------------------------------------------------------------------------
// Stage N contiguous elements into LDS (converting to f16 when needed).
//  - float source: VGPR round-trip is required for v_cvt_pk_f16_f32.
//  - f16 source:   direct global_load_async_to_lds_b128 (no VGPR round-trip).
// ---------------------------------------------------------------------------
template <int N>
__device__ __forceinline__ void stage_chunk(const float* __restrict__ src,
                                            _Float16* __restrict__ dst)
{
    #pragma unroll
    for (int c = 0; c < N; ++c)
        dst[c] = (_Float16)src[c];
}

template <int N>
__device__ __forceinline__ void stage_chunk(const _Float16* __restrict__ src,
                                            _Float16* __restrict__ dst)
{
    #pragma unroll
    for (int i = 0; i < N / 8; ++i)
        async_copy_b128(src + i * 8, dst + i * 8);
}

// ---------------------------------------------------------------------------
// Cooperative stage of one K-chunk (32 wide) into LDS:
//   Ah[64][40]  : A rows (M-major), f16
//   Bh[128][40] : B columns stored N-major (Bh[n][k]), f16
// A: [M,K] row-major.  B: !BT -> [K,N] row-major (register-transposed with
// v4h packed ds_store_b64), BT -> [N,K] row-major (contiguous).
// ---------------------------------------------------------------------------
template <typename TA, typename TB, bool BT>
__device__ __forceinline__ void stage_tiles(
    const TA* __restrict__ A, int lda,
    const TB* __restrict__ B, int ldb,
    int tileM, int tileN, int k0, int tid,
    _Float16 (*__restrict__ Ah)[40], _Float16 (*__restrict__ Bh)[40])
{
    // ---- A: 64 rows x 32 k, 2 threads per row, 16 contiguous elems each ----
    {
        const int arow = tid >> 1;
        const int acs  = (tid & 1) * 16;
        stage_chunk<16>(&A[(size_t)(tileM + arow) * lda + (k0 + acs)],
                        &Ah[arow][acs]);
    }
    // ---- B: 128 n x 32 k ----
    if (BT) {
        // source [N,K] row-major: one thread per n-row, 32 contiguous k
        stage_chunk<32>(&B[(size_t)(tileN + tid) * ldb + k0], &Bh[tid][0]);
    } else {
        // source [K,N] row-major: 4(k) x 8(n) sub-block per thread,
        // transposed in registers, stored as packed v4h (ds_store_b64)
        const int kk0 = (tid >> 4) * 4;   // 0,4,..,28
        const int n0  = (tid & 15) * 8;   // 0,8,..,120
        _Float16 rh[4][8];
        #pragma unroll
        for (int kk = 0; kk < 4; ++kk) {
            const TB* src = &B[(size_t)(k0 + kk0 + kk) * ldb + (tileN + n0)];
            #pragma unroll
            for (int j = 0; j < 8; ++j)
                rh[kk][j] = (_Float16)src[j];
        }
        #pragma unroll
        for (int j = 0; j < 8; ++j) {
            v4h p = {rh[0][j], rh[1][j], rh[2][j], rh[3][j]};
            *reinterpret_cast<v4h*>(&Bh[n0 + j][kk0]) = p;
        }
    }
}

// ---------------------------------------------------------------------------
// 64x128-tile double-buffered WMMA GEMM:  C = alpha * (A*B + bias) + res
// Block = 128 threads = 4 wave32s; each wave owns a 32x64 sub-tile
// (2x4 of 16x16) -> 8 v_wmma_f32_16x16x32_f16 per wave per K-step.
// All dims are multiples of tile sizes here (8192 / 2048 / 1024).
// ---------------------------------------------------------------------------
template <typename TA, typename TB, bool BT, typename TOUT, bool HAS_BIAS, bool HAS_RES>
__global__ __launch_bounds__(128) void wmma_gemm_kernel(
    const TA* __restrict__ A, int lda,
    const TB* __restrict__ B, int ldb,
    const float* __restrict__ bias,
    const float* __restrict__ res, int ldr,
    TOUT* __restrict__ C, int ldc,
    int K, float alpha)
{
    __shared__ alignas(16) _Float16 Ah[2][64][40];    // 10 KB
    __shared__ alignas(16) _Float16 Bh[2][128][40];   // 20 KB

    const int tid  = threadIdx.x;
    const int lane = tid & 31;
    const int wave = tid >> 5;
    const int wm   = wave >> 1;       // wave row 0..1 (32 rows each)
    const int wn   = wave & 1;        // wave col 0..1 (64 cols each)
    const int half = lane >> 4;       // lane group 0/1
    const int l16  = lane & 15;
    const int tileM = blockIdx.y * 64;
    const int tileN = blockIdx.x * 128;

    v8f acc[2][4];
    #pragma unroll
    for (int i = 0; i < 2; ++i)
        #pragma unroll
        for (int j = 0; j < 4; ++j) {
            v8f z = {};
            acc[i][j] = z;
        }

    const int nk = K >> 5;

    // prologue: stage first K-chunk into buffer 0
    stage_tiles<TA, TB, BT>(A, lda, B, ldb, tileM, tileN, 0, tid, Ah[0], Bh[0]);
    wait_async_then_barrier();

    for (int kt = 0; kt < nk; ++kt) {
        const int cur = kt & 1;
        // stage next chunk into the other buffer while we compute (uniform branch)
        if (kt + 1 < nk)
            stage_tiles<TA, TB, BT>(A, lda, B, ldb, tileM, tileN,
                                    (kt + 1) << 5, tid, Ah[cur ^ 1], Bh[cur ^ 1]);

        // ---- fragments per ISA 16-bit A 16x32 layout:
        // lane L -> row L%16; elems 0..7: K=8*(L/16)+e; elems 8..15: K=16+8*(L/16)+e
        const _Float16 (*Ac)[40] = Ah[cur];
        const _Float16 (*Bc)[40] = Bh[cur];

        v16h af[2], bf[4];
        #pragma unroll
        for (int s = 0; s < 2; ++s) {
            const int r = wm * 32 + s * 16 + l16;
            const v8h lo = *reinterpret_cast<const v8h*>(&Ac[r][half * 8]);
            const v8h hi = *reinterpret_cast<const v8h*>(&Ac[r][16 + half * 8]);
            #pragma unroll
            for (int e = 0; e < 8; ++e) { af[s][e] = lo[e]; af[s][8 + e] = hi[e]; }
        }
        #pragma unroll
        for (int t = 0; t < 4; ++t) {
            const int n = wn * 64 + t * 16 + l16;
            const v8h lo = *reinterpret_cast<const v8h*>(&Bc[n][half * 8]);
            const v8h hi = *reinterpret_cast<const v8h*>(&Bc[n][16 + half * 8]);
            #pragma unroll
            for (int e = 0; e < 8; ++e) { bf[t][e] = lo[e]; bf[t][8 + e] = hi[e]; }
        }

        #pragma unroll
        for (int sm = 0; sm < 2; ++sm)
            #pragma unroll
            for (int sn = 0; sn < 4; ++sn)
                acc[sm][sn] = __builtin_amdgcn_wmma_f32_16x16x32_f16(
                    false, af[sm], false, bf[sn],
                    (short)0, acc[sm][sn], false, false);

        wait_async_then_barrier();
    }

    // ---- epilogue: D layout -> lane L holds (M = 8*(L/16)+j , N = L%16) ----
    #pragma unroll
    for (int sm = 0; sm < 2; ++sm) {
        #pragma unroll
        for (int sn = 0; sn < 4; ++sn) {
            const int colg = tileN + wn * 64 + sn * 16 + l16;
            #pragma unroll
            for (int j = 0; j < 8; ++j) {
                const int rowg = tileM + wm * 32 + sm * 16 + half * 8 + j;
                float v = acc[sm][sn][j];
                if (HAS_BIAS) v += bias[colg];
                v *= alpha;
                if (HAS_RES) v += res[(size_t)rowg * ldr + colg];
                C[(size_t)rowg * ldc + colg] = (TOUT)v;
            }
        }
    }
}

// ---------------------------------------------------------------------------
// In-place row softmax over f16 scores. One block (256 thr) per row of 2048.
// ---------------------------------------------------------------------------
__global__ __launch_bounds__(256) void softmax_rows_kernel(_Float16* __restrict__ S, int n)
{
    __shared__ float red[256];
    _Float16* row = S + (size_t)blockIdx.x * n;
    const int tid = threadIdx.x;
    const int base = tid * 8;

    v8h x = *reinterpret_cast<const v8h*>(&row[base]);
    float v[8];
    #pragma unroll
    for (int e = 0; e < 8; ++e) v[e] = (float)x[e];

    float lmax = v[0];
    #pragma unroll
    for (int e = 1; e < 8; ++e) lmax = fmaxf(lmax, v[e]);
    red[tid] = lmax; __syncthreads();
    for (int s = 128; s > 0; s >>= 1) {
        if (tid < s) red[tid] = fmaxf(red[tid], red[tid + s]);
        __syncthreads();
    }
    const float rmax = red[0]; __syncthreads();

    float lsum = 0.f;
    float ev[8];
    #pragma unroll
    for (int e = 0; e < 8; ++e) { ev[e] = __expf(v[e] - rmax); lsum += ev[e]; }
    red[tid] = lsum; __syncthreads();
    for (int s = 128; s > 0; s >>= 1) {
        if (tid < s) red[tid] += red[tid + s];
        __syncthreads();
    }
    const float inv = 1.0f / red[0];

    v8h y;
    #pragma unroll
    for (int e = 0; e < 8; ++e) y[e] = (_Float16)(ev[e] * inv);
    *reinterpret_cast<v8h*>(&row[base]) = y;
}

// ---------------------------------------------------------------------------
// Row LayerNorm over H=1024 f32. One block (256 thr, 4 elems each) per row.
// ---------------------------------------------------------------------------
__global__ __launch_bounds__(256) void layernorm_kernel(
    const float* __restrict__ X, const float* __restrict__ gamma,
    const float* __restrict__ beta, float* __restrict__ out, float eps)
{
    __shared__ float rs[256];
    __shared__ float rq[256];
    const int tid = threadIdx.x;
    const size_t rowoff = (size_t)blockIdx.x * 1024;
    const float4 x = *reinterpret_cast<const float4*>(&X[rowoff + tid * 4]);

    float s = x.x + x.y + x.z + x.w;
    float q = x.x * x.x + x.y * x.y + x.z * x.z + x.w * x.w;
    rs[tid] = s; rq[tid] = q; __syncthreads();
    for (int k = 128; k > 0; k >>= 1) {
        if (tid < k) { rs[tid] += rs[tid + k]; rq[tid] += rq[tid + k]; }
        __syncthreads();
    }
    const float mu   = rs[0] * (1.0f / 1024.0f);
    const float var  = rq[0] * (1.0f / 1024.0f) - mu * mu;
    const float rstd = rsqrtf(var + eps);

    const int c = tid * 4;
    float4 g = *reinterpret_cast<const float4*>(&gamma[c]);
    float4 b = *reinterpret_cast<const float4*>(&beta[c]);
    float4 o;
    o.x = (x.x - mu) * rstd * g.x + b.x;
    o.y = (x.y - mu) * rstd * g.y + b.y;
    o.z = (x.z - mu) * rstd * g.z + b.z;
    o.w = (x.w - mu) * rstd * g.w + b.w;
    *reinterpret_cast<float4*>(&out[rowoff + c]) = o;
}

// ---------------------------------------------------------------------------
extern "C" void kernel_launch(void* const* d_in, const int* in_sizes, int n_in,
                              void* d_out, int out_size, void* d_ws, size_t ws_size,
                              hipStream_t stream)
{
    (void)in_sizes; (void)n_in; (void)out_size; (void)ws_size;

    const float* hidden = (const float*)d_in[0];
    const float* Wq = (const float*)d_in[1];
    const float* bq = (const float*)d_in[2];
    const float* Wk = (const float*)d_in[3];
    const float* bk = (const float*)d_in[4];
    const float* Wv = (const float*)d_in[5];
    const float* bv = (const float*)d_in[6];
    const float* Wd = (const float*)d_in[7];
    const float* bd = (const float*)d_in[8];
    const float* ln_g = (const float*)d_in[9];
    const float* ln_b = (const float*)d_in[10];

    constexpr int Hh = 1024, Bb = 4, Ss = 2048;
    constexpr int M = Bb * Ss;                 // 8192 rows
    constexpr float inv_sqrt_h = 0.03125f;     // 1/sqrt(1024)

    // workspace carve-out (f16 intermediates; scores reused in place as probs)
    _Float16* Qh  = (_Float16*)d_ws;                            // [M,H]
    _Float16* Kh  = Qh + (size_t)M * Hh;                        // [M,H]
    _Float16* Vh  = Kh + (size_t)M * Hh;                        // [M,H]
    _Float16* Sc  = Vh + (size_t)M * Hh;                        // [B,S,S]
    _Float16* Ctx = Sc + (size_t)Bb * Ss * Ss;                  // [M,H]
    float*    X1  = (float*)(Ctx + (size_t)M * Hh);             // [M,H] f32

    const dim3 blk(128);

    // --- QKV projections (f32 inputs -> f16 outputs; Q pre-scaled by 1/sqrt(H)) ---
    const dim3 gProj(Hh / 128, M / 64);   // (8, 128)
    wmma_gemm_kernel<float, float, false, _Float16, true, false>
        <<<gProj, blk, 0, stream>>>(hidden, Hh, Wq, Hh, bq, nullptr, 0, Qh, Hh, Hh, inv_sqrt_h);
    wmma_gemm_kernel<float, float, false, _Float16, true, false>
        <<<gProj, blk, 0, stream>>>(hidden, Hh, Wk, Hh, bk, nullptr, 0, Kh, Hh, Hh, 1.0f);
    wmma_gemm_kernel<float, float, false, _Float16, true, false>
        <<<gProj, blk, 0, stream>>>(hidden, Hh, Wv, Hh, bv, nullptr, 0, Vh, Hh, Hh, 1.0f);

    // --- scores = Q * K^T (per batch); 32 MB f16 stays L2-resident on MI455X ---
    const dim3 gScore(Ss / 128, Ss / 64); // (16, 32)
    for (int b = 0; b < Bb; ++b)
        wmma_gemm_kernel<_Float16, _Float16, true, _Float16, false, false>
            <<<gScore, blk, 0, stream>>>(Qh + (size_t)b * Ss * Hh, Hh,
                                         Kh + (size_t)b * Ss * Hh, Hh,
                                         nullptr, nullptr, 0,
                                         Sc + (size_t)b * Ss * Ss, Ss, Hh, 1.0f);

    // --- softmax in place over each score row ---
    softmax_rows_kernel<<<dim3(Bb * Ss), dim3(256), 0, stream>>>(Sc, Ss);

    // --- ctx = P * V (per batch) ---
    const dim3 gPV(Hh / 128, Ss / 64);    // (8, 32)
    for (int b = 0; b < Bb; ++b)
        wmma_gemm_kernel<_Float16, _Float16, false, _Float16, false, false>
            <<<gPV, blk, 0, stream>>>(Sc + (size_t)b * Ss * Ss, Ss,
                                      Vh + (size_t)b * Ss * Hh, Hh,
                                      nullptr, nullptr, 0,
                                      Ctx + (size_t)b * Ss * Hh, Hh, Ss, 1.0f);

    // --- output projection + bias + residual ---
    wmma_gemm_kernel<_Float16, float, false, float, true, true>
        <<<gProj, blk, 0, stream>>>(Ctx, Hh, Wd, Hh, bd, hidden, Hh, X1, Hh, Hh, 1.0f);

    // --- LayerNorm ---
    layernorm_kernel<<<dim3(M), dim3(256), 0, stream>>>(X1, ln_g, ln_b, (float*)d_out, 1e-5f);
}
